// TModel_24756191494620
// MI455X (gfx1250) — matrix operand order repaired and checked by reading, not compiled, not run on hardware
//
#include <hip/hip_runtime.h>
#include <hip/hip_bf16.h>

typedef __attribute__((ext_vector_type(16))) __bf16 v16bf;
typedef __attribute__((ext_vector_type(8)))  __bf16 v8bf;
typedef __attribute__((ext_vector_type(8)))  float  v8f;
typedef __attribute__((ext_vector_type(4)))  float  v4f;

// ---------- helpers ----------
__device__ __forceinline__ v8bf ld8_f32_to_bf16(const float* __restrict__ p) {
    v4f f0 = *(const v4f*)p;
    v4f f1 = *(const v4f*)(p + 4);
    v8bf h;
#pragma unroll
    for (int i = 0; i < 4; ++i) { h[i] = (__bf16)f0[i]; h[i + 4] = (__bf16)f1[i]; }
    return h;
}

__device__ __forceinline__ v8bf ldb8(const __bf16* p) { return *(const v8bf*)p; }

__device__ __forceinline__ v16bf cat16(v8bf lo, v8bf hi) {
    v16bf r;
#pragma unroll
    for (int i = 0; i < 8; ++i) { r[i] = lo[i]; r[i + 8] = hi[i]; }
    return r;
}

__device__ __forceinline__ float leaky01(float f) { return f > 0.0f ? f : 0.1f * f; }

// ---------- kernel 0: convert weights f32 -> bf16 into workspace ----------
__global__ __launch_bounds__(256) void gnn_prep_weights(
    const float* __restrict__ W1a, const float* __restrict__ W1b,
    const float* __restrict__ W2a, const float* __restrict__ W2b,
    __bf16* __restrict__ w1a, __bf16* __restrict__ w1b,
    __bf16* __restrict__ w2a, __bf16* __restrict__ w2b)
{
    int i = blockIdx.x * 256 + threadIdx.x;            // grid covers 16384
    if (i < 128 * 128) w1a[i] = (__bf16)W1a[i];
    if (i < 128 * 128) w1b[i] = (__bf16)W1b[i];
    if (i < 64 * 256)  w2a[i] = (__bf16)W2a[i];
    if (i < 64 * 64)   w2b[i] = (__bf16)W2b[i];
}

// ---------- kernel 1: edge MLP (two 128x128 layers) + scatter-add ----------
// block = 256 threads (8 waves); each block handles 128 edges; each wave 16 edges.
__global__ __launch_bounds__(256, 2) void gnn_edge_mlp_scatter(
    const float* __restrict__ x_s, const float* __restrict__ edge_attr,
    const int* __restrict__ eidx, long long E,
    const __bf16* __restrict__ W1a, const float* __restrict__ b1a,
    const __bf16* __restrict__ W1b, const float* __restrict__ b1b,
    float* __restrict__ agg)
{
    __shared__ __align__(16) __bf16 sH[128 * 128];   // layer-1 activations (bf16), 32 KB

    const long long e0 = (long long)blockIdx.x * 128;
    const int lane    = threadIdx.x & 31;
    const int wave    = threadIdx.x >> 5;
    const int rowbase = wave << 4;          // 16 rows per wave
    const int lhalf   = lane >> 4;          // 0: K{0-7,16-23}  1: K{8-15,24-31}
    const int ncol    = lane & 15;
    const int aoff    = lhalf << 3;         // 0 or 8

    // ---- A fragments for layer 1: row = e0 + rowbase + ncol, cols = concat(x_s[src], edge_attr) ----
    long long eA = e0 + rowbase + ncol; if (eA >= E) eA = E - 1;
    const int   src  = eidx[eA];
    const float* xrow = x_s       + (size_t)src * 64;
    const float* erow = edge_attr + (size_t)eA  * 64;

    v16bf A1[4];
#pragma unroll
    for (int ks = 0; ks < 4; ++ks) {
        int c0 = ks * 32 + aoff;
        int c1 = c0 + 16;
        v8bf lo = (c0 < 64) ? ld8_f32_to_bf16(xrow + c0) : ld8_f32_to_bf16(erow + (c0 - 64));
        v8bf hi = (c1 < 64) ? ld8_f32_to_bf16(xrow + c1) : ld8_f32_to_bf16(erow + (c1 - 64));
        A1[ks] = cat16(lo, hi);
    }

    // ---- target ids for this lane's 8 output rows (D-layout rows) ----
    int tg[8];
#pragma unroll
    for (int i = 0; i < 8; ++i) {
        long long er = e0 + rowbase + aoff + i;
        tg[i] = (er < E) ? eidx[E + er] : -1;
    }

    // ---- layer 1: sH = leaky(msg @ W1a^T + b1a) ----
#pragma unroll
    for (int nt = 0; nt < 8; ++nt) {
        const int N = (nt << 4) + ncol;
        const float bias = b1a[N];
        v8f acc;
#pragma unroll
        for (int i = 0; i < 8; ++i) acc[i] = bias;
#pragma unroll
        for (int ks = 0; ks < 4; ++ks) {
            const __bf16* bp = W1a + (size_t)N * 128 + ks * 32 + (lhalf << 3);
            v16bf Bv = cat16(ldb8(bp), ldb8(bp + 16));
            acc = __builtin_amdgcn_wmma_f32_16x16x32_bf16(
                false, A1[ks], false, Bv, (short)0, acc, false, false);
        }
#pragma unroll
        for (int i = 0; i < 8; ++i)
            sH[(rowbase + aoff + i) * 128 + N] = (__bf16)leaky01(acc[i]);
    }

    // ---- re-stripe: A fragments for layer 2 (wave-local rows, LDS in-order per wave) ----
    v16bf A2[4];
    const __bf16* hrow = sH + (rowbase + ncol) * 128;
#pragma unroll
    for (int ks = 0; ks < 4; ++ks) {
        int c0 = ks * 32 + aoff;
        A2[ks] = cat16(ldb8(hrow + c0), ldb8(hrow + c0 + 16));
    }

    // ---- layer 2 + scatter-add to agg[tgt] ----
#pragma unroll
    for (int nt = 0; nt < 8; ++nt) {
        const int N = (nt << 4) + ncol;
        const float bias = b1b[N];
        v8f acc;
#pragma unroll
        for (int i = 0; i < 8; ++i) acc[i] = bias;
#pragma unroll
        for (int ks = 0; ks < 4; ++ks) {
            const __bf16* bp = W1b + (size_t)N * 128 + ks * 32 + (lhalf << 3);
            v16bf Bv = cat16(ldb8(bp), ldb8(bp + 16));
            acc = __builtin_amdgcn_wmma_f32_16x16x32_bf16(
                false, A2[ks], false, Bv, (short)0, acc, false, false);
        }
#pragma unroll
        for (int i = 0; i < 8; ++i) {
            if (tg[i] >= 0)
                atomicAdd(&agg[(size_t)tg[i] * 128 + N], acc[i]);
        }
    }
}

// ---------- kernel 2: node MLP  h=[x_t | agg | u[batch]] (256) -> 64 -> 64 ----------
__global__ __launch_bounds__(256, 2) void gnn_node_mlp(
    const float* __restrict__ x_t, const float* __restrict__ agg,
    const float* __restrict__ u,   const int* __restrict__ batch_t,
    const __bf16* __restrict__ W2a, const float* __restrict__ b2a,
    const __bf16* __restrict__ W2b, const float* __restrict__ b2b,
    float* __restrict__ out, int NT)
{
    __shared__ __align__(16) __bf16 sH[128 * 64];   // layer-A activations, 16 KB

    const int r0      = blockIdx.x * 128;
    const int lane    = threadIdx.x & 31;
    const int wave    = threadIdx.x >> 5;
    const int rowbase = wave << 4;
    const int lhalf   = lane >> 4;
    const int ncol    = lane & 15;
    const int aoff    = lhalf << 3;

    // ---- A fragments for layer A: row = r0 + rowbase + ncol, 256 cols ----
    int rA = r0 + rowbase + ncol;
    int rc = (rA < NT) ? rA : (NT - 1);
    const float* xrow = x_t + (size_t)rc * 64;
    const float* arow = agg + (size_t)rc * 128;
    const float* urow = u   + (size_t)batch_t[rc] * 64;

    v16bf A1[8];
#pragma unroll
    for (int ks = 0; ks < 8; ++ks) {
        int c0 = ks * 32 + aoff;
        int c1 = c0 + 16;
        v8bf lo = (c0 < 64)  ? ld8_f32_to_bf16(xrow + c0)
                : (c0 < 192) ? ld8_f32_to_bf16(arow + (c0 - 64))
                             : ld8_f32_to_bf16(urow + (c0 - 192));
        v8bf hi = (c1 < 64)  ? ld8_f32_to_bf16(xrow + c1)
                : (c1 < 192) ? ld8_f32_to_bf16(arow + (c1 - 64))
                             : ld8_f32_to_bf16(urow + (c1 - 192));
        A1[ks] = cat16(lo, hi);
    }

    // ---- layer A: sH = leaky(h @ W2a^T + b2a), out width 64, K=256 ----
#pragma unroll
    for (int nt = 0; nt < 4; ++nt) {
        const int N = (nt << 4) + ncol;
        const float bias = b2a[N];
        v8f acc;
#pragma unroll
        for (int i = 0; i < 8; ++i) acc[i] = bias;
#pragma unroll
        for (int ks = 0; ks < 8; ++ks) {
            const __bf16* bp = W2a + (size_t)N * 256 + ks * 32 + (lhalf << 3);
            v16bf Bv = cat16(ldb8(bp), ldb8(bp + 16));
            acc = __builtin_amdgcn_wmma_f32_16x16x32_bf16(
                false, A1[ks], false, Bv, (short)0, acc, false, false);
        }
#pragma unroll
        for (int i = 0; i < 8; ++i)
            sH[(rowbase + aoff + i) * 64 + N] = (__bf16)leaky01(acc[i]);
    }

    // ---- re-stripe for layer B (K=64) ----
    v16bf A2[2];
    const __bf16* hrow = sH + (rowbase + ncol) * 64;
#pragma unroll
    for (int ks = 0; ks < 2; ++ks) {
        int c0 = ks * 32 + aoff;
        A2[ks] = cat16(ldb8(hrow + c0), ldb8(hrow + c0 + 16));
    }

    // ---- layer B + store ----
#pragma unroll
    for (int nt = 0; nt < 4; ++nt) {
        const int N = (nt << 4) + ncol;
        const float bias = b2b[N];
        v8f acc;
#pragma unroll
        for (int i = 0; i < 8; ++i) acc[i] = bias;
#pragma unroll
        for (int ks = 0; ks < 2; ++ks) {
            const __bf16* bp = W2b + (size_t)N * 64 + ks * 32 + (lhalf << 3);
            v16bf Bv = cat16(ldb8(bp), ldb8(bp + 16));
            acc = __builtin_amdgcn_wmma_f32_16x16x32_bf16(
                false, A2[ks], false, Bv, (short)0, acc, false, false);
        }
#pragma unroll
        for (int i = 0; i < 8; ++i) {
            int r = r0 + rowbase + aoff + i;
            if (r < NT) out[(size_t)r * 64 + N] = acc[i];
        }
    }
}

// ---------- host launcher ----------
extern "C" void kernel_launch(void* const* d_in, const int* in_sizes, int n_in,
                              void* d_out, int out_size, void* d_ws, size_t ws_size,
                              hipStream_t stream) {
    const float* x_s       = (const float*)d_in[0];
    const float* x_t       = (const float*)d_in[1];
    const int*   eidx      = (const int*)  d_in[2];   // [2, E] flattened: src then tgt
    const float* edge_attr = (const float*)d_in[3];
    const float* u         = (const float*)d_in[4];
    const int*   batch_t   = (const int*)  d_in[5];
    const float* W1a = (const float*)d_in[6];
    const float* b1a = (const float*)d_in[7];
    const float* W1b = (const float*)d_in[8];
    const float* b1b = (const float*)d_in[9];
    const float* W2a = (const float*)d_in[10];
    const float* b2a = (const float*)d_in[11];
    const float* W2b = (const float*)d_in[12];
    const float* b2b = (const float*)d_in[13];

    const long long E  = (long long)in_sizes[2] / 2;
    const int       NT = in_sizes[1] / 64;

    // workspace layout: agg f32 [NT*128] | W1a bf16 | W1b bf16 | W2a bf16 | W2b bf16
    const size_t aggBytes = (size_t)NT * 128 * sizeof(float);
    char*   ws  = (char*)d_ws;
    float*  agg = (float*)ws;
    __bf16* w1a = (__bf16*)(ws + aggBytes);
    __bf16* w1b = w1a + 128 * 128;
    __bf16* w2a = w1b + 128 * 128;
    __bf16* w2b = w2a + 64 * 256;

    hipMemsetAsync(agg, 0, aggBytes, stream);

    gnn_prep_weights<<<(128 * 128 + 255) / 256, 256, 0, stream>>>(
        W1a, W1b, W2a, W2b, w1a, w1b, w2a, w2b);

    const int nblk1 = (int)((E + 127) / 128);
    gnn_edge_mlp_scatter<<<nblk1, 256, 0, stream>>>(
        x_s, edge_attr, eidx, E, w1a, b1a, w1b, b1b, agg);

    const int nblk2 = (NT + 127) / 128;
    gnn_node_mlp<<<nblk2, 256, 0, stream>>>(
        x_t, agg, u, batch_t, w2a, b2a, w2b, b2b, (float*)d_out, NT);
}
